// STHGNNLayerIndividual_26225070309935
// MI455X (gfx1250) — matrix operand order repaired and checked by reading, not compiled, not run on hardware
//
#include <hip/hip_runtime.h>

// ---------------------------------------------------------------------------
// Problem constants (B=32, T=12, N=1024, C=64, NW=9, 4N=4096, 2F=128)
// ---------------------------------------------------------------------------
typedef __attribute__((ext_vector_type(16))) __bf16 v16bf;
typedef __attribute__((ext_vector_type(8)))  float  v8f;
typedef __attribute__((ext_vector_type(4)))  int    v4i;

union U16 { unsigned int u[8]; v16bf v; };

#if defined(__has_builtin)
#if __has_builtin(__builtin_amdgcn_global_load_async_to_lds_b128) && \
    __has_builtin(__builtin_amdgcn_s_wait_asynccnt)
#define USE_ASYNC 1
#endif
#endif
#ifndef USE_ASYNC
#define USE_ASYNC 0
#endif

__device__ inline unsigned short f2bf(float f) {
  unsigned int u = __builtin_bit_cast(unsigned int, f);
  u += 0x7fffu + ((u >> 16) & 1u);          // round-to-nearest-even
  return (unsigned short)(u >> 16);
}

__device__ inline float sigmoidf(float x) { return 1.0f / (1.0f + __expf(-x)); }

// A-fragment (16x32 bf16, contiguous k = 16*half + s) from row-major [row][k] LDS tile.
__device__ inline v16bf ld_afrag(const unsigned short* row, int half, int kbase) {
  U16 t;
#pragma unroll
  for (int j = 0; j < 8; ++j) {
    int k = kbase + 2 * j + 16 * half;
    t.u[j] = *(const unsigned int*)(row + k);
  }
  return t.v;
}

// B-fragment (32x16 bf16) from a [n][k] LDS tile (K pairs contiguous).
__device__ inline v16bf ld_bfrag(const unsigned short* col, int half, int kbase) {
  U16 t;
#pragma unroll
  for (int j = 0; j < 8; ++j)
    t.u[j] = *(const unsigned int*)(col + kbase + 2 * j + 16 * half);
  return t.v;
}

// B-fragment (32x16 bf16) from a row-major [k][c] tile (pitch 72), paired u16.
__device__ inline v16bf ld_bfrag_kc(const unsigned short (*tile)[72], int col,
                                    int half) {
  U16 t;
#pragma unroll
  for (int j = 0; j < 8; ++j) {
    int k0 = 2 * j + 16 * half;
    unsigned int lo = tile[k0][col];
    unsigned int hi = tile[k0 + 1][col];
    t.u[j] = lo | (hi << 16);
  }
  return t.v;
}

#if USE_ASYNC
__device__ inline void async_cp16(const void* g, void* l) {
  __builtin_amdgcn_global_load_async_to_lds_b128(
      (__attribute__((address_space(1))) v4i*)g,
      (__attribute__((address_space(3))) v4i*)l, 0, 0);
}
#endif

// ---------------------------------------------------------------------------
// prep: x = data + temporal_emb + spatial_emb -> bf16   [32,12,1024,64]
// ---------------------------------------------------------------------------
__global__ __launch_bounds__(256) void prep_x(const float4* __restrict__ data,
                                              const float4* __restrict__ te,
                                              const float4* __restrict__ se,
                                              ushort4* __restrict__ xb) {
  int idx = blockIdx.x * 256 + threadIdx.x;      // float4 index
  if (idx >= 6291456) return;                    // 32*12*1024*16
  int c4 = idx & 15;
  int n  = (idx >> 4) & 1023;
  int t  = (idx >> 14) % 12;
  float4 d = data[idx], a = te[t * 16 + c4], s = se[n * 16 + c4];
  ushort4 o;
  o.x = f2bf(d.x + a.x + s.x);
  o.y = f2bf(d.y + a.y + s.y);
  o.z = f2bf(d.z + a.z + s.z);
  o.w = f2bf(d.w + a.w + s.w);
  xb[idx] = o;
}

__global__ __launch_bounds__(256) void prep_adj(const float4* __restrict__ adj,
                                                ushort4* __restrict__ adjb) {
  int idx = blockIdx.x * 256 + threadIdx.x;      // float4 index
  if (idx >= 4194304) return;                    // 4096*4096/4
  float4 d = adj[idx];
  ushort4 o;
  o.x = f2bf(d.x); o.y = f2bf(d.y); o.z = f2bf(d.z); o.w = f2bf(d.w);
  adjb[idx] = o;
}

// ---------------------------------------------------------------------------
// Gated dilated temporal conv via WMMA (writes data_res into d_out)
// ---------------------------------------------------------------------------
__global__ __launch_bounds__(256) void conv_kernel(
    const unsigned short* __restrict__ xb, const float* __restrict__ wl,
    const float* __restrict__ bl, const float* __restrict__ wr,
    const float* __restrict__ br, float* __restrict__ dout) {
  __shared__ unsigned short sX[128][40];    // x tile [row][k-chunk of 32]
  __shared__ unsigned short sW[128][136];   // W''^T  [o][k 0..127]
  int tid = threadIdx.x;
  int wv = tid >> 5, lane = tid & 31, half = lane >> 4, m = lane & 15;
  int nbase = blockIdx.x * 128;
  int bt = blockIdx.y, b = bt / 9, t = bt % 9;

  { // stage weights transposed: W''[k][o]: k<64 -> tap0, else tap1
    int o = tid >> 1, ks = (tid & 1) * 64;
    const float* wsrc = (o < 64) ? wl : wr;
    int oo = o & 63;
#pragma unroll 8
    for (int k = 0; k < 64; ++k) {
      int kk = ks + k, c = kk & 63, j = kk >> 6;
      sW[o][kk] = f2bf(wsrc[(oo * 64 + c) * 2 + j]);
    }
  }
  __syncthreads();

  v8f acc[8];
#pragma unroll
  for (int i = 0; i < 8; ++i) acc[i] = v8f{0.f, 0.f, 0.f, 0.f, 0.f, 0.f, 0.f, 0.f};

  for (int kk = 0; kk < 128; kk += 32) {
    { // stage 128x32 x-tile (contiguous channels at time t or t+3)
      int r = tid >> 1, sg = (tid & 1) * 16;
      const unsigned short* src = xb +
          ((size_t)(b * 12 + t + (kk >= 64 ? 3 : 0)) * 1024 + nbase + r) * 64 +
          (kk & 63) + sg;
      uint4 p0 = *(const uint4*)(src);
      uint4 p1 = *(const uint4*)(src + 8);
      unsigned int* d = (unsigned int*)&sX[r][sg];
      d[0] = p0.x; d[1] = p0.y; d[2] = p0.z; d[3] = p0.w;
      d[4] = p1.x; d[5] = p1.y; d[6] = p1.z; d[7] = p1.w;
    }
    __syncthreads();
    v16bf a = ld_afrag(&sX[wv * 16 + m][0], half, 0);
#pragma unroll
    for (int ot = 0; ot < 8; ++ot) {
      v16bf bb = ld_bfrag(&sW[ot * 16 + m][0], half, kk);
      acc[ot] = __builtin_amdgcn_wmma_f32_16x16x32_bf16(
          false, a, false, bb, (short)0, acc[ot], false, false);
    }
    __syncthreads();
  }

  int nrow = nbase + wv * 16;
#pragma unroll
  for (int ot = 0; ot < 4; ++ot) {
    float bL = bl[ot * 16 + m];
    float bR = br[ot * 16 + m];
#pragma unroll
    for (int r = 0; r < 8; ++r) {
      float l = acc[ot][r] + bL;
      float g = acc[ot + 4][r] + bR;
      int row = nrow + r + 8 * half;
      dout[(((size_t)b * 9 + t) * 1024 + row) * 64 + ot * 16 + m] =
          sigmoidf(l) * tanhf(g);
    }
  }
}

// ---------------------------------------------------------------------------
// One STSGCM hop:  h' = GLU( (adj @ h) @ W_k + b_k )
// Block tile: 256 v-rows x 2 batches x 64 cols, double-buffered LDS staging.
// LDS aliasing: stage1 [sA 2x256x40 | sB 2x2x32x72] 59,392 B
//               stage2 [sG 256x72 | sW 128x72]      55,296 B
// grid: (v-tiles, 16 batch-pairs)
// ---------------------------------------------------------------------------
#define SA_OFF(p)      ((p) * 20480)
#define SB_OFF(p, bb)  (40960 + (p) * 9216 + (bb) * 4608)
#define SG_OFF         0
#define SW_OFF         36864

__device__ inline void stage_tiles(unsigned char* smem, int p, int kk, int tid,
                                   const unsigned short* __restrict__ adjb,
                                   size_t vrow0,
                                   const unsigned short* __restrict__ hb0,
                                   const unsigned short* __restrict__ hb1) {
  // adj tile: one 256x32 chunk, one 32-elem row per thread
  unsigned short (*sA)[40] = (unsigned short(*)[40])(smem + SA_OFF(p));
  const unsigned short* ap = adjb + (vrow0 + tid) * 4096 + kk;
  __builtin_prefetch(ap + 32, 0, 1);              // global_prefetch_b8
  // h tiles: 2 batches x 32k x 64c row-major, 16 elems per thread
  int bb = tid >> 7, kr = (tid >> 2) & 31, cs = (tid & 3) * 16;
  const unsigned short* hp = (bb ? hb1 : hb0) + (size_t)(kk + kr) * 64 + cs;
  unsigned short (*sBt)[72] = (unsigned short(*)[72])(smem + SB_OFF(p, bb));
#if USE_ASYNC
#pragma unroll
  for (int i = 0; i < 4; ++i) async_cp16(ap + 8 * i, &sA[tid][8 * i]);
  async_cp16(hp, &sBt[kr][cs]);
  async_cp16(hp + 8, &sBt[kr][cs + 8]);
#else
  uint4 a0 = *(const uint4*)(ap);
  uint4 a1 = *(const uint4*)(ap + 8);
  uint4 a2 = *(const uint4*)(ap + 16);
  uint4 a3 = *(const uint4*)(ap + 24);
  uint4* da = (uint4*)&sA[tid][0];
  da[0] = a0; da[1] = a1; da[2] = a2; da[3] = a3;
  uint4 q0 = *(const uint4*)(hp);
  uint4 q1 = *(const uint4*)(hp + 8);
  uint4* db = (uint4*)&sBt[kr][cs];
  db[0] = q0; db[1] = q1;
#endif
}

__device__ inline void stage_fence() {
#if USE_ASYNC
  __builtin_amdgcn_s_wait_asynccnt(0);
#endif
  __syncthreads();
}

__global__ __launch_bounds__(256) void agg_kernel(
    const unsigned short* __restrict__ adjb,
    const unsigned short* __restrict__ h_in, long hin_bstride,
    unsigned short* __restrict__ h_out,
    const float* __restrict__ fcw, const float* __restrict__ fcb,
    float* __restrict__ smax, float* __restrict__ dout,
    int kidx, int vbase0, int win) {
  __shared__ __align__(16) unsigned char smem[59392];
  int tid = threadIdx.x;
  int wv = tid >> 5, lane = tid & 31, half = lane >> 4, m = lane & 15;
  int vbase = vbase0 + blockIdx.x * 256;
  int b0 = blockIdx.y * 2;
  const unsigned short* hb0 = h_in + (size_t)b0 * hin_bstride;
  const unsigned short* hb1 = hb0 + hin_bstride;

  v8f acc[16];  // [rf(2)][bb(2)][ct(4)]
#pragma unroll
  for (int i = 0; i < 16; ++i)
    acc[i] = v8f{0.f, 0.f, 0.f, 0.f, 0.f, 0.f, 0.f, 0.f};

  stage_tiles(smem, 0, 0, tid, adjb, (size_t)vbase, hb0, hb1);
  stage_fence();

  int p = 0;
  for (int kk = 0; kk < 4096; kk += 32) {
    if (kk + 32 < 4096)
      stage_tiles(smem, p ^ 1, kk + 32, tid, adjb, (size_t)vbase, hb0, hb1);

    const unsigned short (*sA)[40] = (const unsigned short(*)[40])(smem + SA_OFF(p));
    v16bf a0 = ld_afrag(&sA[wv * 32 + m][0], half, 0);
    v16bf a1 = ld_afrag(&sA[wv * 32 + 16 + m][0], half, 0);
#pragma unroll
    for (int bb = 0; bb < 2; ++bb) {
      const unsigned short (*sBt)[72] =
          (const unsigned short(*)[72])(smem + SB_OFF(p, bb));
#pragma unroll
      for (int ct = 0; ct < 4; ++ct) {
        v16bf bf = ld_bfrag_kc(sBt, ct * 16 + m, half);
        acc[(0 * 2 + bb) * 4 + ct] = __builtin_amdgcn_wmma_f32_16x16x32_bf16(
            false, a0, false, bf, (short)0, acc[(0 * 2 + bb) * 4 + ct], false, false);
        acc[(1 * 2 + bb) * 4 + ct] = __builtin_amdgcn_wmma_f32_16x16x32_bf16(
            false, a1, false, bf, (short)0, acc[(1 * 2 + bb) * 4 + ct], false, false);
      }
    }
    stage_fence();
    p ^= 1;
  }

  // ---- stage 2: g = agg @ W_k, GLU, outputs (aliased LDS region) ----
  unsigned short (*sG)[72] = (unsigned short(*)[72])(smem + SG_OFF);
  unsigned short (*sW)[72] = (unsigned short(*)[72])(smem + SW_OFF);
  { // W_k transposed: fcw is [c=64][o=128] row-major
    int o = tid >> 1, cs2 = (tid & 1) * 32;
#pragma unroll 8
    for (int c2 = 0; c2 < 32; ++c2)
      sW[o][cs2 + c2] = f2bf(fcw[(size_t)(cs2 + c2) * 128 + o]);
  }

  bool isMid = (vbase >= 1024) && (vbase < 2048);

  for (int bb = 0; bb < 2; ++bb) {
    int b = b0 + bb;
#pragma unroll
    for (int rf = 0; rf < 2; ++rf)
#pragma unroll
      for (int ct = 0; ct < 4; ++ct)
#pragma unroll
        for (int r = 0; r < 8; ++r)
          sG[wv * 32 + rf * 16 + r + 8 * half][ct * 16 + m] =
              f2bf(acc[(rf * 2 + bb) * 4 + ct][r]);
    __syncthreads();

#pragma unroll
    for (int rf = 0; rf < 2; ++rf) {
      v8f g[8];
#pragma unroll
      for (int i = 0; i < 8; ++i)
        g[i] = v8f{0.f, 0.f, 0.f, 0.f, 0.f, 0.f, 0.f, 0.f};
#pragma unroll
      for (int ks = 0; ks < 64; ks += 32) {
        v16bf a = ld_afrag(&sG[wv * 32 + rf * 16 + m][0], half, ks);
#pragma unroll
        for (int ot = 0; ot < 8; ++ot) {
          v16bf bf = ld_bfrag(&sW[ot * 16 + m][0], half, ks);
          g[ot] = __builtin_amdgcn_wmma_f32_16x16x32_bf16(
              false, a, false, bf, (short)0, g[ot], false, false);
        }
      }
#pragma unroll
      for (int ot = 0; ot < 4; ++ot) {
        float bF = fcb[ot * 16 + m];
        float bG = fcb[64 + ot * 16 + m];
#pragma unroll
        for (int r = 0; r < 8; ++r) {
          float val = (g[ot][r] + bF) * sigmoidf(g[ot + 4][r] + bG);  // GLU
          int v = vbase + wv * 32 + rf * 16 + r + 8 * half;
          int f = ot * 16 + m;
          if (kidx < 2) {
            h_out[((size_t)b * 4096 + v) * 64 + f] = f2bf(val);
            if (isMid) {
              size_t si = ((size_t)b * 1024 + (v - 1024)) * 64 + f;
              smax[si] = (kidx == 0) ? val : fmaxf(smax[si], val);
            }
          } else {
            size_t si = ((size_t)b * 1024 + (v - 1024)) * 64 + f;
            size_t oi = (((size_t)b * 9 + win) * 1024 + (v - 1024)) * 64 + f;
            dout[oi] = dout[oi] + fmaxf(smax[si], val);
          }
        }
      }
    }
    __syncthreads();   // before sG is overwritten for next batch
  }
}

// ---------------------------------------------------------------------------
extern "C" void kernel_launch(void* const* d_in, const int* in_sizes, int n_in,
                              void* d_out, int out_size, void* d_ws, size_t ws_size,
                              hipStream_t stream) {
  (void)in_sizes; (void)n_in; (void)out_size; (void)ws_size;
  const float* data = (const float*)d_in[0];
  const float* adj  = (const float*)d_in[1];
  const float* temb = (const float*)d_in[2];
  const float* semb = (const float*)d_in[3];
  const float* cwl  = (const float*)d_in[4];
  const float* cbl  = (const float*)d_in[5];
  const float* cwr  = (const float*)d_in[6];
  const float* cbr  = (const float*)d_in[7];
  const float* fcw  = (const float*)d_in[8];
  const float* fcb  = (const float*)d_in[9];
  float* out = (float*)d_out;

  char* ws = (char*)d_ws;                         // 120 MB layout
  unsigned short* xb   = (unsigned short*)(ws + 0);          // 50,331,648 B
  unsigned short* adjb = (unsigned short*)(ws + 50331648);   // 33,554,432 B
  unsigned short* hA   = (unsigned short*)(ws + 83886080);   // 16,777,216 B
  unsigned short* hB   = (unsigned short*)(ws + 100663296);  // 16,777,216 B
  float*          smax = (float*)(ws + 117440512);           //  8,388,608 B

  prep_x<<<24576, 256, 0, stream>>>((const float4*)data, (const float4*)temb,
                                    (const float4*)semb, (ushort4*)xb);
  prep_adj<<<16384, 256, 0, stream>>>((const float4*)adj, (ushort4*)adjb);
  conv_kernel<<<dim3(8, 288), 256, 0, stream>>>(xb, cwl, cbl, cwr, cbr, out);

  for (int win = 0; win < 9; ++win) {
    const float* W0 = fcw + (size_t)(win * 3 + 0) * 8192;
    const float* W1 = fcw + (size_t)(win * 3 + 1) * 8192;
    const float* W2 = fcw + (size_t)(win * 3 + 2) * 8192;
    const float* b0 = fcb + (size_t)(win * 3 + 0) * 128;
    const float* b1 = fcb + (size_t)(win * 3 + 1) * 128;
    const float* b2 = fcb + (size_t)(win * 3 + 2) * 128;
    // k=0: h0 is the contiguous slab x[b, win:win+4] (batch stride 12*1024*64)
    agg_kernel<<<dim3(16, 16), 256, 0, stream>>>(adjb, xb + (size_t)win * 65536,
                                                 786432L, hA, W0, b0, smax, out,
                                                 0, 0, win);
    agg_kernel<<<dim3(16, 16), 256, 0, stream>>>(adjb, hA, 262144L, hB, W1, b1,
                                                 smax, out, 1, 0, win);
    // k=2: only the middle vertex block (rows 1024..2047) is needed
    agg_kernel<<<dim3(4, 16), 256, 0, stream>>>(adjb, hB, 262144L, nullptr, W2,
                                                b2, smax, out, 2, 1024, win);
  }
}
